// SequentialTransductionUnitJagged_17231408792410
// MI455X (gfx1250) — matrix operand order repaired and checked by reading, not compiled, not run on hardware
//
#include <hip/hip_runtime.h>

// ---------------------------------------------------------------------------
// Types
// ---------------------------------------------------------------------------
typedef unsigned short ushort_t;
typedef __attribute__((ext_vector_type(16))) __bf16        v16bf;
typedef __attribute__((ext_vector_type(8)))  float         v8f;
typedef __attribute__((ext_vector_type(4)))  unsigned int  u32x4;

union V16U { v16bf b; u32x4 q[2]; ushort_t s[16]; };
union V8U  { u32x4 q;  ushort_t s[8]; };

#define D_EMB   512
#define N_HEAD  8
#define D_HEAD  64
#define N_MASK  1024
#define LN_EPS  1e-6f

// ---------------------------------------------------------------------------
// Small device helpers
// ---------------------------------------------------------------------------
__device__ __forceinline__ ushort_t f2bf(float f) {
    unsigned u = __float_as_uint(f);
    unsigned r = u + 0x7FFFu + ((u >> 16) & 1u);   // round-to-nearest-even
    return (ushort_t)(r >> 16);
}
__device__ __forceinline__ float bf2f(ushort_t h) {
    return __uint_as_float(((unsigned)h) << 16);
}
__device__ __forceinline__ float silu(float x) {
    // x * rcp(1 + exp(-x)) : v_exp_f32 + v_rcp_f32, no IEEE divide chain
    return x * __builtin_amdgcn_rcpf(1.0f + __expf(-x));
}

__device__ __forceinline__ v16bf frag_ld32(const ushort_t* p) {
    V16U u;
    u.q[0] = *reinterpret_cast<const u32x4*>(p);
    u.q[1] = *reinterpret_cast<const u32x4*>(p + 8);
    return u.b;
}
__device__ __forceinline__ v16bf frag_ld2(const ushort_t* p0, const ushort_t* p1) {
    V16U u;
    u.q[0] = *reinterpret_cast<const u32x4*>(p0);
    u.q[1] = *reinterpret_cast<const u32x4*>(p1);
    return u.b;
}
__device__ __forceinline__ v16bf frag_zero() {
    V16U u;
    u.q[0] = (u32x4)0u;
    u.q[1] = (u32x4)0u;
    return u.b;
}
__device__ __forceinline__ v8f wmma_bf16(v16bf a, v16bf b, v8f c) {
    // D = A(16x32 bf16) * B(32x16 bf16) + C(16x16 f32)
    return __builtin_amdgcn_wmma_f32_16x16x32_bf16(false, a, false, b, (short)0, c,
                                                   false, false);
}

// CDNA5 async global -> LDS copy (16B per lane), ASYNCcnt-tracked (GV mode).
// Low 32 bits of a generic shared-memory pointer == LDS byte offset.
__device__ __forceinline__ void async_ld_b128(void* lds_ptr, const void* gptr) {
    unsigned lds = (unsigned)(unsigned long long)lds_ptr;
    asm volatile("global_load_async_to_lds_b128 %0, %1, off"
                 :: "v"(lds), "v"(gptr) : "memory");
}
__device__ __forceinline__ void wait_async0() {
    asm volatile("s_wait_asynccnt 0x0" ::: "memory");
}

// ---------------------------------------------------------------------------
// LayerNorm of x (f32, row = 512) -> bf16. One block (256 thr) per token row.
// ---------------------------------------------------------------------------
__global__ __launch_bounds__(256) void ln_x_kernel(const float* __restrict__ x,
                                                   ushort_t* __restrict__ xn) {
    __shared__ float red[256];
    const int t   = blockIdx.x;
    const int tid = threadIdx.x;
    const float* row = x + (size_t)t * D_EMB;
    float v0 = row[tid];
    float v1 = row[tid + 256];

    red[tid] = v0 + v1;
    __syncthreads();
    for (int o = 128; o > 0; o >>= 1) {
        if (tid < o) red[tid] += red[tid + o];
        __syncthreads();
    }
    float mean = red[0] * (1.0f / (float)D_EMB);
    __syncthreads();

    float d0 = v0 - mean, d1 = v1 - mean;
    red[tid] = d0 * d0 + d1 * d1;
    __syncthreads();
    for (int o = 128; o > 0; o >>= 1) {
        if (tid < o) red[tid] += red[tid + o];
        __syncthreads();
    }
    float rstd = rsqrtf(red[0] * (1.0f / (float)D_EMB) + LN_EPS);

    xn[(size_t)t * D_EMB + tid]       = f2bf(d0 * rstd);
    xn[(size_t)t * D_EMB + tid + 256] = f2bf(d1 * rstd);
}

// ---------------------------------------------------------------------------
// o_input = u * layernorm(attn) -> bf16
// ---------------------------------------------------------------------------
__global__ __launch_bounds__(256) void ln_mul_kernel(const ushort_t* __restrict__ attn,
                                                     const float* __restrict__ u,
                                                     ushort_t* __restrict__ oin) {
    __shared__ float red[256];
    const int t   = blockIdx.x;
    const int tid = threadIdx.x;
    const ushort_t* row = attn + (size_t)t * D_EMB;
    float v0 = bf2f(row[tid]);
    float v1 = bf2f(row[tid + 256]);

    red[tid] = v0 + v1;
    __syncthreads();
    for (int o = 128; o > 0; o >>= 1) {
        if (tid < o) red[tid] += red[tid + o];
        __syncthreads();
    }
    float mean = red[0] * (1.0f / (float)D_EMB);
    __syncthreads();

    float d0 = v0 - mean, d1 = v1 - mean;
    red[tid] = d0 * d0 + d1 * d1;
    __syncthreads();
    for (int o = 128; o > 0; o >>= 1) {
        if (tid < o) red[tid] += red[tid + o];
        __syncthreads();
    }
    float rstd = rsqrtf(red[0] * (1.0f / (float)D_EMB) + LN_EPS);

    size_t base = (size_t)t * D_EMB;
    oin[base + tid]       = f2bf(u[base + tid] * (d0 * rstd));
    oin[base + tid + 256] = f2bf(u[base + tid + 256] * (d1 * rstd));
}

// ---------------------------------------------------------------------------
// f32 -> bf16 convert, and o_weight transpose+convert
// ---------------------------------------------------------------------------
__global__ __launch_bounds__(256) void cvt_bf16_kernel(const float* __restrict__ in,
                                                       ushort_t* __restrict__ out, int n) {
    int i = blockIdx.x * 256 + threadIdx.x;
    if (i < n) out[i] = f2bf(in[i]);
}

__global__ __launch_bounds__(256) void owt_kernel(const float* __restrict__ w,
                                                  ushort_t* __restrict__ wT) {
    // wT[c*512 + d] = w[d*512 + c]   (so GEMM3 B is row-major K x N)
    int i = blockIdx.x * 256 + threadIdx.x;   // i over 512*512
    int c = i >> 9;
    int d = i & 511;
    wT[i] = f2bf(w[(size_t)d * D_EMB + c]);
}

// ---------------------------------------------------------------------------
// Shared WMMA GEMM main loop: 128x128 tile, K in steps of 32, 8 waves.
//   A: row-major M x 512 bf16  (lda = 512)  -> LDS via async global->LDS DMA
//   B: row-major 512 x N bf16  (ldb param)  -> LDS transposed via VGPR path
// Wave (w): rows (w>>1)*32 .. +32, cols (w&1)*64 .. +64 -> acc[2][4] 16x16 tiles
// ---------------------------------------------------------------------------
__device__ __forceinline__ void gemm_mainloop(const ushort_t* __restrict__ A,
                                              const ushort_t* __restrict__ B,
                                              int M, int ldb, int row0, int n0,
                                              ushort_t* aS, ushort_t* bT,
                                              v8f acc[2][4]) {
    const int tid  = threadIdx.x;
    const int wave = tid >> 5;
    const int lane = tid & 31;
    const int hi   = lane >> 4;
    const int lm   = lane & 15;
    const int mw   = (wave >> 1) * 32;
    const int nw   = (wave & 1) * 64;

    for (int mt = 0; mt < 2; ++mt)
        for (int nt = 0; nt < 4; ++nt)
            acc[mt][nt] = (v8f)0.0f;

    const int ar  = tid >> 1;             // A: row within tile
    const int ach = (tid & 1) * 16;       // A: 16-elem column chunk
    const int bkr = tid >> 3;             // B: k-row within tile
    const int bc  = (tid & 7) * 16;       // B: 16-elem column chunk

    for (int k0 = 0; k0 < D_EMB; k0 += 32) {
        __syncthreads();
        // ---- A tile 128x32 row-major: async global->LDS (1x16B per lane) ----
        if (row0 + ar < M) {
            const ushort_t* src = A + (size_t)(row0 + ar) * D_EMB + k0 + ach;
            async_ld_b128(aS + ar * 32 + ach, src);
        } else {
            *reinterpret_cast<u32x4*>(aS + ar * 32 + ach)     = (u32x4)0u;
            *reinterpret_cast<u32x4*>(aS + ar * 32 + ach + 8) = (u32x4)0u;
        }
        // ---- B tile 32x128, store transposed [n][k] ----
        {
            const ushort_t* src = B + (size_t)(k0 + bkr) * ldb + n0 + bc;
            V16U u;
            u.q[0] = *reinterpret_cast<const u32x4*>(src);
            u.q[1] = *reinterpret_cast<const u32x4*>(src + 8);
#pragma unroll
            for (int j = 0; j < 16; ++j) bT[(bc + j) * 32 + bkr] = u.s[j];
        }
        // prefetch next K-step tiles while this one is consumed
        if (k0 + 32 < D_EMB) {
            __builtin_prefetch(A + (size_t)(row0 + ar) * D_EMB + k0 + 32 + ach, 0, 0);
            __builtin_prefetch(B + (size_t)(k0 + 32 + bkr) * ldb + n0 + bc, 0, 0);
        }
        wait_async0();
        __syncthreads();

        // ---- fragments + 8 WMMAs ----
        v16bf af[2], bfg[4];
#pragma unroll
        for (int mt = 0; mt < 2; ++mt) {
            const ushort_t* arp = aS + (mw + mt * 16 + lm) * 32;
            af[mt] = frag_ld32(arp + 16 * hi);
        }
#pragma unroll
        for (int nt = 0; nt < 4; ++nt)
            bfg[nt] = frag_ld32(bT + (nw + nt * 16 + lm) * 32 + 16 * hi);
#pragma unroll
        for (int mt = 0; mt < 2; ++mt)
#pragma unroll
            for (int nt = 0; nt < 4; ++nt)
                acc[mt][nt] = wmma_bf16(af[mt], bfg[nt], acc[mt][nt]);
    }
}

// ---------------------------------------------------------------------------
// GEMM1: mm = silu(xn @ uvqk_bf16), split columns into u(f32), v, q, k (bf16)
// ---------------------------------------------------------------------------
__global__ __launch_bounds__(256) void gemm1_kernel(const ushort_t* __restrict__ xn,
                                                    const ushort_t* __restrict__ w,
                                                    float* __restrict__ u_out,
                                                    ushort_t* __restrict__ v_out,
                                                    ushort_t* __restrict__ q_out,
                                                    ushort_t* __restrict__ k_out,
                                                    int M) {
    __shared__ ushort_t aS[128 * 32];
    __shared__ ushort_t bT[128 * 32];
    const int row0 = blockIdx.x * 128;
    const int n0   = blockIdx.y * 128;

    v8f acc[2][4];
    gemm_mainloop(xn, w, M, 4 * D_EMB, row0, n0, aS, bT, acc);

    const int lane = threadIdx.x & 31;
    const int wave = threadIdx.x >> 5;
    const int hi   = lane >> 4;
    const int lm   = lane & 15;
    const int mw   = (wave >> 1) * 32;
    const int nw   = (wave & 1) * 64;

#pragma unroll
    for (int mt = 0; mt < 2; ++mt)
#pragma unroll
        for (int nt = 0; nt < 4; ++nt)
#pragma unroll
            for (int r = 0; r < 8; ++r) {
                int grow = row0 + mw + mt * 16 + 8 * hi + r;
                if (grow >= M) continue;
                int gcol = n0 + nw + nt * 16 + lm;
                float s  = silu(acc[mt][nt][r]);
                int seg  = gcol >> 9;
                size_t idx = (size_t)grow * D_EMB + (gcol & 511);
                if (seg == 0)      u_out[idx] = s;
                else if (seg == 1) v_out[idx] = f2bf(s);
                else if (seg == 2) q_out[idx] = f2bf(s);
                else               k_out[idx] = f2bf(s);
            }
}

// ---------------------------------------------------------------------------
// GEMM3: out = oin @ owT + bias + x   (f32 output, residual)
// ---------------------------------------------------------------------------
__global__ __launch_bounds__(256) void gemm3_kernel(const ushort_t* __restrict__ oin,
                                                    const ushort_t* __restrict__ wT,
                                                    const float* __restrict__ bias,
                                                    const float* __restrict__ x,
                                                    float* __restrict__ out,
                                                    int M) {
    __shared__ ushort_t aS[128 * 32];
    __shared__ ushort_t bT[128 * 32];
    const int row0 = blockIdx.x * 128;
    const int n0   = blockIdx.y * 128;

    v8f acc[2][4];
    gemm_mainloop(oin, wT, M, D_EMB, row0, n0, aS, bT, acc);

    const int lane = threadIdx.x & 31;
    const int wave = threadIdx.x >> 5;
    const int hi   = lane >> 4;
    const int lm   = lane & 15;
    const int mw   = (wave >> 1) * 32;
    const int nw   = (wave & 1) * 64;

#pragma unroll
    for (int mt = 0; mt < 2; ++mt)
#pragma unroll
        for (int nt = 0; nt < 4; ++nt)
#pragma unroll
            for (int r = 0; r < 8; ++r) {
                int grow = row0 + mw + mt * 16 + 8 * hi + r;
                if (grow >= M) continue;
                int gcol = n0 + nw + nt * 16 + lm;
                size_t idx = (size_t)grow * D_EMB + gcol;
                out[idx] = acc[mt][nt][r] + bias[gcol] + x[idx];
            }
}

// ---------------------------------------------------------------------------
// Jagged causal HSTU attention.
// grid.x = B*H, grid.y = q-tile (128 rows). 8 waves, each wave = 16 q rows x 64 DV.
// K block -> LDS via async DMA (row-major feeds S-WMMA B-frags directly),
// V block -> LDS transposed via VGPR path. S = Q.K^T (4 WMMAs), silu/N +
// causal mask, P via per-wave LDS transpose, O += P.V (4 WMMAs).
// ---------------------------------------------------------------------------
__global__ __launch_bounds__(256) void attn_kernel(const ushort_t* __restrict__ qb,
                                                   const ushort_t* __restrict__ kbuf,
                                                   const ushort_t* __restrict__ vbuf,
                                                   ushort_t* __restrict__ attn,
                                                   const int* __restrict__ offs) {
    __shared__ ushort_t kS[32 * 64];     // [key][d]
    __shared__ ushort_t vT[64 * 32];     // [dv][key]
    __shared__ ushort_t pS[8 * 16 * 32]; // per-wave P staging [m][key]

    const int bh  = blockIdx.x;
    const int b   = bh >> 3;
    const int h   = bh & 7;
    const int off = offs[b];
    const int len = offs[b + 1] - off;
    const int qbase = blockIdx.y * 128;
    if (qbase >= len) return;

    const int tid  = threadIdx.x;
    const int wave = tid >> 5;
    const int lane = tid & 31;
    const int hi   = lane >> 4;
    const int lm   = lane & 15;
    const int m0   = qbase + wave * 16;     // first q row (in-seq) of this wave

    // ---- Q fragments (2 k-steps over DA=64), loaded straight from global ----
    v16bf qa[2];
    {
        int rq = m0 + lm;
        bool valid = rq < len;
        const ushort_t* qp = qb + (size_t)(off + rq) * D_EMB + h * D_HEAD;
#pragma unroll
        for (int ks = 0; ks < 2; ++ks)
            qa[ks] = valid ? frag_ld32(qp + ks * 32 + 16 * hi)
                           : frag_zero();
    }

    v8f acc[4];
#pragma unroll
    for (int nt = 0; nt < 4; ++nt) acc[nt] = (v8f)0.0f;

    const float inv_n = 1.0f / (float)N_MASK;
    const int kmax = (len < qbase + 128) ? len : (qbase + 128);
    const int key  = tid >> 3;
    const int c8   = (tid & 7) * 8;

    for (int kb = 0; kb < kmax; kb += 32) {
        __syncthreads();
        // ---- cooperative K/V block load (32 keys x 64) ----
        {
            size_t rowb = (size_t)(off + kb + key) * D_EMB + h * D_HEAD + c8;
            if (kb + key < len) {
                // K: async global->LDS DMA, row-major, no VGPR round-trip
                async_ld_b128(kS + key * 64 + c8, kbuf + rowb);
                // V: VGPR path with transposed LDS store
                V8U vd;
                vd.q = *reinterpret_cast<const u32x4*>(vbuf + rowb);
#pragma unroll
                for (int j = 0; j < 8; ++j) vT[(c8 + j) * 32 + key] = vd.s[j];
            } else {
                *reinterpret_cast<u32x4*>(kS + key * 64 + c8) = (u32x4)0u;
#pragma unroll
                for (int j = 0; j < 8; ++j) vT[(c8 + j) * 32 + key] = 0;
            }
        }
        wait_async0();
        __syncthreads();

        // ---- S = Q.K^T : two 16x16 key tiles ----
        v8f s0 = (v8f)0.0f, s1 = (v8f)0.0f;
        s0 = wmma_bf16(qa[0], frag_ld32(kS + lm * 64 + 16 * hi), s0);
        s0 = wmma_bf16(qa[1], frag_ld32(kS + lm * 64 + 32 + 16 * hi), s0);
        s1 = wmma_bf16(qa[0], frag_ld32(kS + (16 + lm) * 64 + 16 * hi), s1);
        s1 = wmma_bf16(qa[1], frag_ld32(kS + (16 + lm) * 64 + 32 + 16 * hi), s1);

        // ---- silu / N * causal mask, stage P (bf16) through per-wave LDS ----
        ushort_t* pw = pS + wave * (16 * 32);
        const int j0 = kb + lm;
        const int j1 = kb + 16 + lm;
#pragma unroll
        for (int r = 0; r < 8; ++r) {
            int m = m0 + 8 * hi + r;             // query row (in-seq)
            float a0 = silu(s0[r]) * inv_n;
            float a1 = silu(s1[r]) * inv_n;
            a0 = (j0 <= m) ? a0 : 0.0f;
            a1 = (j1 <= m) ? a1 : 0.0f;
            pw[(8 * hi + r) * 32 + lm]      = f2bf(a0);
            pw[(8 * hi + r) * 32 + 16 + lm] = f2bf(a1);
        }
        // cross-lane transpose through LDS: wave-internal, DS is in-order,
        // but force ordering vs the compiler + hardware counters explicitly
        asm volatile("s_wait_dscnt 0" ::: "memory");

        v16bf pa = frag_ld32(pw + lm * 32 + 16 * hi);
#pragma unroll
        for (int nt = 0; nt < 4; ++nt)
            acc[nt] = wmma_bf16(pa, frag_ld32(vT + (nt * 16 + lm) * 32 + 16 * hi),
                                acc[nt]);
    }

    // ---- store attn rows (jagged) ----
#pragma unroll
    for (int nt = 0; nt < 4; ++nt)
#pragma unroll
        for (int r = 0; r < 8; ++r) {
            int m = m0 + 8 * hi + r;
            if (m < len)
                attn[(size_t)(off + m) * D_EMB + h * D_HEAD + nt * 16 + lm] =
                    f2bf(acc[nt][r]);
        }
}

// ---------------------------------------------------------------------------
// Host launcher
// ---------------------------------------------------------------------------
extern "C" void kernel_launch(void* const* d_in, const int* in_sizes, int n_in,
                              void* d_out, int out_size, void* d_ws, size_t ws_size,
                              hipStream_t stream) {
    const float* x      = (const float*)d_in[0];
    const int*   offs   = (const int*)d_in[1];
    // d_in[2] = invalid_attn_mask (causal tril, constant) -> computed analytically
    const float* uvqk   = (const float*)d_in[3];
    const float* ow     = (const float*)d_in[4];
    const float* ob     = (const float*)d_in[5];
    float*       out    = (float*)d_out;

    const int T = in_sizes[0] / D_EMB;           // total jagged tokens
    const int B = in_sizes[1] - 1;               // number of jagged sequences

    // ---- workspace layout (bf16 stored as ushort) ----
    char* ws = (char*)d_ws;
    size_t tb = (size_t)T * D_EMB;
    ushort_t* xn      = (ushort_t*)ws;            ws += tb * 2;
    float*    u_buf   = (float*)ws;               ws += tb * 4;
    ushort_t* v_buf   = (ushort_t*)ws;            ws += tb * 2;
    ushort_t* q_buf   = (ushort_t*)ws;            ws += tb * 2;
    ushort_t* k_buf   = (ushort_t*)ws;            ws += tb * 2;
    ushort_t* attn_b  = (ushort_t*)ws;            ws += tb * 2;
    ushort_t* oin     = (ushort_t*)ws;            ws += tb * 2;
    ushort_t* uvqk_bf = (ushort_t*)ws;            ws += (size_t)D_EMB * 4 * D_EMB * 2;
    ushort_t* owT     = (ushort_t*)ws;            ws += (size_t)D_EMB * D_EMB * 2;

    const int mtiles = (T + 127) / 128;

    // 1) layernorm(x) -> bf16
    ln_x_kernel<<<T, 256, 0, stream>>>(x, xn);
    // 2) weight converts
    cvt_bf16_kernel<<<(D_EMB * 4 * D_EMB + 255) / 256, 256, 0, stream>>>(
        uvqk, uvqk_bf, D_EMB * 4 * D_EMB);
    owt_kernel<<<(D_EMB * D_EMB) / 256, 256, 0, stream>>>(ow, owT);
    // 3) projection GEMM + silu + split (u f32, v/q/k bf16)
    gemm1_kernel<<<dim3(mtiles, 16), 256, 0, stream>>>(xn, uvqk_bf, u_buf, v_buf,
                                                       q_buf, k_buf, T);
    // 4) jagged causal attention
    attn_kernel<<<dim3(B * N_HEAD, N_MASK / 128), 256, 0, stream>>>(
        q_buf, k_buf, v_buf, attn_b, offs);
    // 5) o_input = u * layernorm(attn)
    ln_mul_kernel<<<T, 256, 0, stream>>>(attn_b, u_buf, oin);
    // 6) output GEMM + bias + residual
    gemm3_kernel<<<dim3(mtiles, 4), 256, 0, stream>>>(oin, owT, ob, x, out, T);
}
